// TransformerBlock_7447473291696
// MI455X (gfx1250) — matrix operand order repaired and checked by reading, not compiled, not run on hardware
//
#include <hip/hip_runtime.h>

// ---------------------------------------------------------------------------
// Types for WMMA fragments (CDNA5 / gfx1250, wave32)
// ---------------------------------------------------------------------------
typedef __attribute__((ext_vector_type(16))) __bf16        v16bf;
typedef __attribute__((ext_vector_type(8)))  float         v8f;
typedef __attribute__((ext_vector_type(4)))  unsigned int  u32x4;

union FragB {              // 16 bf16 = 8 VGPRs (A or B operand)
    v16bf v;
    u32x4 q[2];
    unsigned short s[16];
};
union FragF {              // 8 f32 = 8 VGPRs (C/D accumulator)
    v8f   v;
    float f[8];
};

__device__ __forceinline__ v8f wmma_bf16(const v16bf& a, const v16bf& b, const v8f& c) {
    // D = A(16x32 bf16) * B(32x16 bf16) + C(16x16 f32)
    return __builtin_amdgcn_wmma_f32_16x16x32_bf16(
        /*neg_a=*/false, a, /*neg_b=*/false, b,
        /*c_mod=*/(short)0, c, /*reuse_a=*/false, /*reuse_b=*/false);
}

__device__ __forceinline__ unsigned short f2bf(float f) {
    unsigned int u = __float_as_uint(f);
    unsigned int r = u + 0x7FFFu + ((u >> 16) & 1u);   // round-to-nearest-even
    return (unsigned short)(r >> 16);
}

// ---------------------------------------------------------------------------
// Model dims
// ---------------------------------------------------------------------------
#define BATCH   2
#define SEQ     2048
#define ROWS    (BATCH * SEQ)     // 4096
#define DMODEL  1024
#define H3      3072
#define DHID    4096
#define NHEADS  16
#define HDIM    64

// ---------------------------------------------------------------------------
// fp32 [K][N] -> bf16 transposed [N][K], 32x32 LDS tile transpose
// ---------------------------------------------------------------------------
__global__ __launch_bounds__(256)
void transpose_cvt_kernel(const float* __restrict__ in,
                          unsigned short* __restrict__ out, int K, int N) {
    __shared__ float tile[32][33];
    const int tx = threadIdx.x & 31;
    const int ty = threadIdx.x >> 5;      // 0..7
    const int k0 = blockIdx.y * 32;
    const int n0 = blockIdx.x * 32;
#pragma unroll
    for (int j = 0; j < 32; j += 8)
        tile[ty + j][tx] = in[(size_t)(k0 + ty + j) * N + n0 + tx];
    __syncthreads();
#pragma unroll
    for (int j = 0; j < 32; j += 8)
        out[(size_t)(n0 + ty + j) * K + k0 + tx] = f2bf(tile[tx][ty + j]);
}

// ---------------------------------------------------------------------------
// LayerNorm over D=1024, one block (256 thr = 8 waves) per row, bf16 output
// ---------------------------------------------------------------------------
__global__ __launch_bounds__(256)
void ln_bf16_kernel(const float* __restrict__ x,
                    const float* __restrict__ g,
                    const float* __restrict__ b,
                    unsigned short* __restrict__ out) {
    const int row = blockIdx.x;
    const int tid = threadIdx.x;
    const float* xr = x + (size_t)row * DMODEL;

    float v[4], s = 0.f, s2 = 0.f;
#pragma unroll
    for (int i = 0; i < 4; ++i) {
        v[i] = xr[tid + i * 256];
        s += v[i];
        s2 += v[i] * v[i];
    }
#pragma unroll
    for (int m = 1; m < 32; m <<= 1) {
        s  += __shfl_xor(s,  m, 32);
        s2 += __shfl_xor(s2, m, 32);
    }
    __shared__ float ps[8], ps2[8];
    const int wid = tid >> 5, lane = tid & 31;
    if (lane == 0) { ps[wid] = s; ps2[wid] = s2; }
    __syncthreads();
    float ts = 0.f, ts2 = 0.f;
#pragma unroll
    for (int w = 0; w < 8; ++w) { ts += ps[w]; ts2 += ps2[w]; }
    const float mu  = ts * (1.0f / DMODEL);
    const float var = ts2 * (1.0f / DMODEL) - mu * mu;
    const float rs  = rsqrtf(var + 1e-5f);

#pragma unroll
    for (int i = 0; i < 4; ++i) {
        int col = tid + i * 256;
        float y = (v[i] - mu) * rs * g[col] + b[col];
        out[(size_t)row * DMODEL + col] = f2bf(y);
    }
}

// ---------------------------------------------------------------------------
// Tiled WMMA GEMM: C[M,N] = A[M,K](bf16) @ Bt[N,K](bf16, pre-transposed) + bias
// Epilogues:
//   EPI_BF16 : out bf16
//   EPI_RESID: out f32 = acc + bias + resid
//   EPI_GELU : out bf16 = gelu_exact(acc + bias)
//   EPI_QKV  : cols < 2048 (Q,K) -> outB; cols >= 2048 (V) -> vtb transposed
//              as [b][col-2048][token] for direct attention B-fragments.
// Block tile 128x128, 256 threads = 8 waves (4 M x 2 N), wave tile 32x64.
// Double-buffered LDS; both operands staged with contiguous b128 ld/st.
// Fragment reads: A lane holds k = {half*16..+15} (16 contiguous),
// B lane holds k = {half*16..+15} (16 contiguous).
// ---------------------------------------------------------------------------
#define EPI_BF16  0
#define EPI_RESID 1
#define EPI_GELU  2
#define EPI_QKV   3
#define LSTR 40

template <int EPI>
__global__ __launch_bounds__(256)
void gemm_bf16_wmma(const unsigned short* __restrict__ A,
                    const unsigned short* __restrict__ Bt,
                    const float* __restrict__ bias,
                    const float* __restrict__ resid,
                    float* __restrict__ outF,
                    unsigned short* __restrict__ outB,
                    unsigned short* __restrict__ vtb,
                    int M, int N, int K) {
    __shared__ unsigned short As[2][128 * LSTR];
    __shared__ unsigned short Bs[2][128 * LSTR];

    const int tid   = threadIdx.x;
    const int lane  = tid & 31;
    const int wid   = tid >> 5;
    const int half  = lane >> 4;
    const int l15   = lane & 15;
    const int waveM = wid & 3;      // 0..3
    const int waveN = wid >> 2;     // 0..1
    const int m0    = blockIdx.y * 128;
    const int n0    = blockIdx.x * 128;

    // staging coords: 512 chunks of 8 bf16 (2 per thread), identical A / Bt
    int srow[2], skc[2];
#pragma unroll
    for (int j = 0; j < 2; ++j) {
        int ci = tid + j * 256;
        srow[j] = ci >> 2;
        skc[j]  = (ci & 3) * 8;
    }

    FragF acc[2][4];
#pragma unroll
    for (int mt = 0; mt < 2; ++mt)
#pragma unroll
        for (int nt = 0; nt < 4; ++nt)
#pragma unroll
            for (int v = 0; v < 8; ++v) acc[mt][nt].f[v] = 0.0f;

    // ---- preload K-tile 0 ----
    u32x4 ra[2], rb[2];
#pragma unroll
    for (int j = 0; j < 2; ++j) {
        ra[j] = *(const u32x4*)(A  + (size_t)(m0 + srow[j]) * K + skc[j]);
        rb[j] = *(const u32x4*)(Bt + (size_t)(n0 + srow[j]) * K + skc[j]);
    }
#pragma unroll
    for (int j = 0; j < 2; ++j) {
        *(u32x4*)&As[0][srow[j] * LSTR + skc[j]] = ra[j];
        *(u32x4*)&Bs[0][srow[j] * LSTR + skc[j]] = rb[j];
    }
    __syncthreads();

    const int KT = K >> 5;
    for (int kt = 0; kt < KT; ++kt) {
        const int cur = kt & 1;
        // ---- issue global loads for next tile (overlaps WMMA burst) ----
        if (kt + 1 < KT) {
            int kb = (kt + 1) << 5;
#pragma unroll
            for (int j = 0; j < 2; ++j) {
                ra[j] = *(const u32x4*)(A  + (size_t)(m0 + srow[j]) * K + kb + skc[j]);
                rb[j] = *(const u32x4*)(Bt + (size_t)(n0 + srow[j]) * K + kb + skc[j]);
            }
            if (kt + 2 < KT) {
                __builtin_prefetch(A  + (size_t)(m0 + srow[0]) * K + kb + 32 + skc[0], 0, 1);
                __builtin_prefetch(Bt + (size_t)(n0 + srow[0]) * K + kb + 32 + skc[0], 0, 1);
            }
        }
        // ---- fragments from current buffer (contiguous ds_load_b128) ----
        FragB aF[2], bF[4];
#pragma unroll
        for (int mt = 0; mt < 2; ++mt) {
            int row = waveM * 32 + mt * 16 + l15;
            aF[mt].q[0] = *(const u32x4*)&As[cur][row * LSTR + half * 16];
            aF[mt].q[1] = *(const u32x4*)&As[cur][row * LSTR + half * 16 + 8];
        }
#pragma unroll
        for (int nt = 0; nt < 4; ++nt) {
            int n = waveN * 64 + nt * 16 + l15;
            bF[nt].q[0] = *(const u32x4*)&Bs[cur][n * LSTR + half * 16];
            bF[nt].q[1] = *(const u32x4*)&Bs[cur][n * LSTR + half * 16 + 8];
        }
#pragma unroll
        for (int mt = 0; mt < 2; ++mt)
#pragma unroll
            for (int nt = 0; nt < 4; ++nt)
                acc[mt][nt].v = wmma_bf16(aF[mt].v, bF[nt].v, acc[mt][nt].v);

        // ---- stash next tile into the other buffer ----
        if (kt + 1 < KT) {
#pragma unroll
            for (int j = 0; j < 2; ++j) {
                *(u32x4*)&As[1 - cur][srow[j] * LSTR + skc[j]] = ra[j];
                *(u32x4*)&Bs[1 - cur][srow[j] * LSTR + skc[j]] = rb[j];
            }
        }
        __syncthreads();
    }

    // ---- epilogue ----
#pragma unroll
    for (int mt = 0; mt < 2; ++mt) {
#pragma unroll
        for (int nt = 0; nt < 4; ++nt) {
            const int colb = n0 + waveN * 64 + nt * 16;   // tile base col (uniform)
            const int col  = colb + l15;
            const float bv = bias[col];
            if (EPI == EPI_QKV && colb >= 2 * DMODEL) {
                // V columns -> vtb[(b*1024 + col') * SEQ + token]
                const int colp = col - 2 * DMODEL;
#pragma unroll
                for (int v = 0; v < 8; ++v) {
                    int row = m0 + waveM * 32 + mt * 16 + half * 8 + v;
                    int bb = row >> 11, t = row & (SEQ - 1);
                    float val = acc[mt][nt].f[v] + bv;
                    vtb[((size_t)(bb * DMODEL + colp)) * SEQ + t] = f2bf(val);
                }
            } else {
#pragma unroll
                for (int v = 0; v < 8; ++v) {
                    int row = m0 + waveM * 32 + mt * 16 + half * 8 + v;
                    size_t idx = (size_t)row * N + col;
                    float val = acc[mt][nt].f[v] + bv;
                    if (EPI == EPI_RESID) {
                        outF[idx] = val + resid[idx];
                    } else if (EPI == EPI_GELU) {
                        val = 0.5f * val * (1.0f + erff(val * 0.70710678f));
                        outB[idx] = f2bf(val);
                    } else { // EPI_BF16 / EPI_QKV (Q,K region)
                        outB[idx] = f2bf(val);
                    }
                }
            }
        }
    }
}

// ---------------------------------------------------------------------------
// Flash attention: one wave per (batch, head, 16-row Q block), 64-key blocks.
// Q,K read from qkv bf16 [ROWS,3072] (head-contiguous 64 cols -> contiguous
// fragments). V read from vtb bf16 [b][h*64+d][token] -> B-fragments are
// contiguous global b128 loads, no LDS staging.
// ---------------------------------------------------------------------------
#define PSTR 72   // P tile LDS row stride (16 rows x 64 keys, padded)

__global__ __launch_bounds__(32)
void attn_fa_kernel(const unsigned short* __restrict__ qkv,
                    const unsigned short* __restrict__ vtb,
                    unsigned short* __restrict__ ctx) {
    __shared__ unsigned short Pl[16 * PSTR];

    const int bid = blockIdx.x;
    const int qi  = bid & 127;           // 128 q-blocks of 16
    const int h   = (bid >> 7) & 15;
    const int b   = bid >> 11;
    const int lane = threadIdx.x;
    const int half = lane >> 4;
    const int l15  = lane & 15;
    const size_t RS = H3;                // qkv row stride (elements)

    // ---- Q fragments (A operand, 16x64 = 2 k-steps of 32) ----
    FragB aQ[2];
    {
        size_t t = (size_t)(b * SEQ + qi * 16 + l15) * RS + h * HDIM;
#pragma unroll
        for (int s = 0; s < 2; ++s) {
            aQ[s].q[0] = *(const u32x4*)(qkv + t + s * 32 + half * 16);
            aQ[s].q[1] = *(const u32x4*)(qkv + t + s * 32 + half * 16 + 8);
        }
    }

    float m8[8], l8[8];
    FragF acc[4];
#pragma unroll
    for (int v = 0; v < 8; ++v) { m8[v] = -3.0e38f; l8[v] = 0.0f; }
#pragma unroll
    for (int nt = 0; nt < 4; ++nt)
#pragma unroll
        for (int v = 0; v < 8; ++v) acc[nt].f[v] = 0.0f;

    const unsigned short* vbase = vtb + (size_t)(b * DMODEL + h * HDIM) * SEQ;

    for (int kb = 0; kb < SEQ / 64; ++kb) {
        // ---- S(16x64) = Q @ K^T : 4 n-tiles x 2 k-steps ----
        FragF sacc[4];
#pragma unroll
        for (int nt = 0; nt < 4; ++nt) {
#pragma unroll
            for (int v = 0; v < 8; ++v) sacc[nt].f[v] = 0.0f;
            size_t tk = (size_t)(b * SEQ + kb * 64 + nt * 16 + l15) * RS + DMODEL + h * HDIM;
#pragma unroll
            for (int s = 0; s < 2; ++s) {
                FragB bK;
                bK.q[0] = *(const u32x4*)(qkv + tk + s * 32 + half * 16);
                bK.q[1] = *(const u32x4*)(qkv + tk + s * 32 + half * 16 + 8);
                sacc[nt].v = wmma_bf16(aQ[s].v, bK.v, sacc[nt].v);
            }
        }

        // ---- online softmax (row = half*8 + v, spread over 16 lanes) ----
        float tmp[8];
#pragma unroll
        for (int v = 0; v < 8; ++v) {
            sacc[0].f[v] *= 0.125f;      // 1/sqrt(64)
            sacc[1].f[v] *= 0.125f;
            sacc[2].f[v] *= 0.125f;
            sacc[3].f[v] *= 0.125f;
            tmp[v] = fmaxf(fmaxf(sacc[0].f[v], sacc[1].f[v]),
                           fmaxf(sacc[2].f[v], sacc[3].f[v]));
        }
#pragma unroll
        for (int msk = 1; msk < 16; msk <<= 1)
#pragma unroll
            for (int v = 0; v < 8; ++v)
                tmp[v] = fmaxf(tmp[v], __shfl_xor(tmp[v], msk, 32));

        float cr[8], rs[8];
#pragma unroll
        for (int v = 0; v < 8; ++v) {
            float nm = fmaxf(m8[v], tmp[v]);
            cr[v] = __expf(m8[v] - nm);
            m8[v] = nm;
            float p0 = __expf(sacc[0].f[v] - nm);
            float p1 = __expf(sacc[1].f[v] - nm);
            float p2 = __expf(sacc[2].f[v] - nm);
            float p3 = __expf(sacc[3].f[v] - nm);
            sacc[0].f[v] = p0; sacc[1].f[v] = p1;
            sacc[2].f[v] = p2; sacc[3].f[v] = p3;
            rs[v] = (p0 + p1) + (p2 + p3);
        }
#pragma unroll
        for (int msk = 1; msk < 16; msk <<= 1)
#pragma unroll
            for (int v = 0; v < 8; ++v)
                rs[v] += __shfl_xor(rs[v], msk, 32);
#pragma unroll
        for (int v = 0; v < 8; ++v) l8[v] = l8[v] * cr[v] + rs[v];
#pragma unroll
        for (int nt = 0; nt < 4; ++nt)
#pragma unroll
            for (int v = 0; v < 8; ++v) acc[nt].f[v] *= cr[v];

        // ---- P (C layout) -> LDS -> A layout ----
#pragma unroll
        for (int nt = 0; nt < 4; ++nt)
#pragma unroll
            for (int v = 0; v < 8; ++v)
                Pl[(half * 8 + v) * PSTR + nt * 16 + l15] = f2bf(sacc[nt].f[v]);
        __syncthreads();

        FragB aP[2];
#pragma unroll
        for (int s = 0; s < 2; ++s) {
            aP[s].q[0] = *(const u32x4*)&Pl[l15 * PSTR + s * 32 + half * 16];
            aP[s].q[1] = *(const u32x4*)&Pl[l15 * PSTR + s * 32 + half * 16 + 8];
        }

        // ---- ctx(16x64) += P(16x64) @ V(64x64), V straight from global ----
#pragma unroll
        for (int s = 0; s < 2; ++s) {
#pragma unroll
            for (int nt = 0; nt < 4; ++nt) {
                FragB bV;
                const unsigned short* vp =
                    vbase + (size_t)(nt * 16 + l15) * SEQ + kb * 64 + s * 32 + half * 16;
                bV.q[0] = *(const u32x4*)(vp);
                bV.q[1] = *(const u32x4*)(vp + 8);
                acc[nt].v = wmma_bf16(aP[s].v, bV.v, acc[nt].v);
            }
        }
        __syncthreads();
    }

    // ---- normalize + store ctx (bf16, token-major, head-contiguous cols) ----
#pragma unroll
    for (int v = 0; v < 8; ++v) {
        float il = 1.0f / l8[v];
        int token = b * SEQ + qi * 16 + half * 8 + v;
#pragma unroll
        for (int nt = 0; nt < 4; ++nt)
            ctx[(size_t)token * DMODEL + h * HDIM + nt * 16 + l15] =
                f2bf(acc[nt].f[v] * il);
    }
}

// ---------------------------------------------------------------------------
// Launch
// ---------------------------------------------------------------------------
extern "C" void kernel_launch(void* const* d_in, const int* in_sizes, int n_in,
                              void* d_out, int out_size, void* d_ws, size_t ws_size,
                              hipStream_t stream) {
    const float* x      = (const float*)d_in[0];
    const float* ln1_g  = (const float*)d_in[1];
    const float* ln1_b  = (const float*)d_in[2];
    const float* ln2_g  = (const float*)d_in[3];
    const float* ln2_b  = (const float*)d_in[4];
    const float* w_qkv  = (const float*)d_in[5];
    const float* b_qkv  = (const float*)d_in[6];
    const float* w_proj = (const float*)d_in[7];
    const float* b_proj = (const float*)d_in[8];
    const float* w_fc1  = (const float*)d_in[9];
    const float* b_fc1  = (const float*)d_in[10];
    const float* w_fc2  = (const float*)d_in[11];
    const float* b_fc2  = (const float*)d_in[12];
    float* out = (float*)d_out;

    // ---- workspace carve (bytes) ----
    unsigned char* w = (unsigned char*)d_ws;
    size_t off = 0;
    auto carve = [&](size_t bytes) {
        unsigned char* p = w + off;
        off += (bytes + 255) & ~(size_t)255;
        return p;
    };
    unsigned short* hb     = (unsigned short*)carve((size_t)ROWS * DMODEL * 2);
    unsigned short* qkvb   = (unsigned short*)carve((size_t)ROWS * H3 * 2);
    unsigned short* vtb    = (unsigned short*)carve((size_t)ROWS * DMODEL * 2);
    unsigned short* ctxb   = (unsigned short*)carve((size_t)ROWS * DMODEL * 2);
    float*          xmid   = (float*)carve((size_t)ROWS * DMODEL * 4);
    unsigned short* h2b    = (unsigned short*)carve((size_t)ROWS * DHID * 2);
    unsigned short* wqkvt  = (unsigned short*)carve((size_t)DMODEL * H3 * 2);
    unsigned short* wprojt = (unsigned short*)carve((size_t)DMODEL * DMODEL * 2);
    unsigned short* wfc1t  = (unsigned short*)carve((size_t)DMODEL * DHID * 2);
    unsigned short* wfc2t  = (unsigned short*)carve((size_t)DHID * DMODEL * 2);
    (void)ws_size; (void)n_in; (void)in_sizes; (void)out_size;

    // 1) weights -> bf16 transposed [N][K]
    transpose_cvt_kernel<<<dim3(H3 / 32, DMODEL / 32), 256, 0, stream>>>(w_qkv, wqkvt, DMODEL, H3);
    transpose_cvt_kernel<<<dim3(DMODEL / 32, DMODEL / 32), 256, 0, stream>>>(w_proj, wprojt, DMODEL, DMODEL);
    transpose_cvt_kernel<<<dim3(DHID / 32, DMODEL / 32), 256, 0, stream>>>(w_fc1, wfc1t, DMODEL, DHID);
    transpose_cvt_kernel<<<dim3(DMODEL / 32, DHID / 32), 256, 0, stream>>>(w_fc2, wfc2t, DHID, DMODEL);

    // 2) LN1 -> bf16
    ln_bf16_kernel<<<ROWS, 256, 0, stream>>>(x, ln1_g, ln1_b, hb);

    // 3) QKV = hb @ wqkv + b_qkv  (Q,K -> qkvb; V transposed -> vtb)
    gemm_bf16_wmma<EPI_QKV><<<dim3(H3 / 128, ROWS / 128), 256, 0, stream>>>(
        hb, wqkvt, b_qkv, nullptr, nullptr, qkvb, vtb, ROWS, H3, DMODEL);

    // 4) attention -> ctx bf16
    attn_fa_kernel<<<BATCH * NHEADS * (SEQ / 16), 32, 0, stream>>>(qkvb, vtb, ctxb);

    // 5) xmid = x + ctx @ w_proj + b_proj   (f32)
    gemm_bf16_wmma<EPI_RESID><<<dim3(DMODEL / 128, ROWS / 128), 256, 0, stream>>>(
        ctxb, wprojt, b_proj, x, xmid, nullptr, nullptr, ROWS, DMODEL, DMODEL);

    // 6) LN2 -> bf16 (reuse hb)
    ln_bf16_kernel<<<ROWS, 256, 0, stream>>>(xmid, ln2_g, ln2_b, hb);

    // 7) h2 = gelu(hb @ w_fc1 + b_fc1)  -> bf16
    gemm_bf16_wmma<EPI_GELU><<<dim3(DHID / 128, ROWS / 128), 256, 0, stream>>>(
        hb, wfc1t, b_fc1, nullptr, nullptr, h2b, nullptr, ROWS, DHID, DMODEL);

    // 8) out = xmid + h2 @ w_fc2 + b_fc2  (f32)
    gemm_bf16_wmma<EPI_RESID><<<dim3(DMODEL / 128, ROWS / 128), 256, 0, stream>>>(
        h2b, wfc2t, b_fc2, xmid, out, nullptr, nullptr, ROWS, DMODEL, DHID);
}
